// BitwiseTasNetRepeat_77644418777120
// MI455X (gfx1250) — 8-bit matrix operand order repaired and checked by reading, compiled, not run on hardware
//
#include <hip/hip_runtime.h>
#include <stdint.h>


typedef int v8i __attribute__((ext_vector_type(8)));

#define T_LEN   8000
#define N_BATCH 4
#define B_CH    128
#define H_CH    512

__device__ inline v8i load_bfrag(const int8_t* __restrict__ sbase, int C,
                                 int trow, int khalf) {
  const int8_t* p = sbase + (size_t)trow * C + khalf * 8;
  int2 q0 = *(const int2*)(p);
  int2 q1 = *(const int2*)(p + 16);
  int2 q2 = *(const int2*)(p + 32);
  int2 q3 = *(const int2*)(p + 48);
  v8i b;
  b[0] = q0.x; b[1] = q0.y; b[2] = q1.x; b[3] = q1.y;
  b[4] = q2.x; b[5] = q2.y; b[6] = q3.x; b[7] = q3.y;
  return b;
}

__device__ inline v8i load_afrag(const int8_t* __restrict__ w, int K,
                                 int row, int kbase, int koff) {
  const int8_t* p = w + (size_t)row * K + kbase + koff;
  int2 q0 = *(const int2*)(p);
  int2 q1 = *(const int2*)(p + 16);
  int2 q2 = *(const int2*)(p + 32);
  int2 q3 = *(const int2*)(p + 48);
  v8i a;
  a[0] = q0.x; a[1] = q0.y; a[2] = q1.x; a[3] = q1.y;
  a[4] = q2.x; a[5] = q2.y; a[6] = q3.x; a[7] = q3.y;
  return a;
}

__global__ __launch_bounds__(256) void k_sign(const float* __restrict__ x,
                                              int8_t* __restrict__ out, long n) {
  long i = (long)blockIdx.x * blockDim.x + threadIdx.x;
  if (i >= n) return;
  float v = x[i];
  out[i] = (v > 0.f) ? (int8_t)1 : ((v < 0.f) ? (int8_t)-1 : (int8_t)0);
}

__global__ __launch_bounds__(256) void k_t_in(const float* __restrict__ x,
                                              float* __restrict__ out) {
  const int C = B_CH, T = T_LEN;
  long total = (long)N_BATCH * T * C;
  long i = (long)blockIdx.x * blockDim.x + threadIdx.x;
  if (i >= total) return;
  int c = (int)(i & (C - 1));
  long nt = i >> 7;
  int t = (int)(nt % T);
  int n = (int)(nt / T);
  out[i] = x[((size_t)n * C + c) * T + t];
}

__global__ __launch_bounds__(256) void k_t_out(const float* __restrict__ r,
                                               float* __restrict__ out) {
  const int C = B_CH, T = T_LEN;
  long total = (long)N_BATCH * C * T;
  long i = (long)blockIdx.x * blockDim.x + threadIdx.x;
  if (i >= total) return;
  int t = (int)(i % T);
  long nb = i / T;
  int b = (int)(nb & (C - 1));
  int n = (int)(nb >> 7);
  out[i] = r[((size_t)n * T + t) * C + b];
}

__device__ inline void block_reduce2(double& s, double& s2) {
  __shared__ double sh[256];
  __shared__ double sh2[256];
  int tid = threadIdx.x;
  sh[tid] = s; sh2[tid] = s2;
  __syncthreads();
  for (int off = 128; off > 0; off >>= 1) {
    if (tid < off) { sh[tid] += sh[tid + off]; sh2[tid] += sh2[tid + off]; }
    __syncthreads();
  }
  s = sh[0]; s2 = sh2[0];
}

template <typename TI>
__global__ __launch_bounds__(256) void k_stats_ntc(
    const TI* __restrict__ x, const float* __restrict__ g,
    const float* __restrict__ bt, float* __restrict__ thr, int C, long NT) {
  int c = blockIdx.x;
  double s = 0.0, s2 = 0.0;
  for (long j = threadIdx.x; j < NT; j += blockDim.x) {
    double v = (double)x[j * C + c];
    s += v; s2 += v * v;
  }
  block_reduce2(s, s2);
  if (threadIdx.x == 0) {
    double cnt = (double)NT;
    double m = s / cnt;
    double var = s2 / cnt - m * m;
    thr[c] = (float)(m - (double)bt[c] * sqrt(var + 1e-5) / (double)g[c]);
  }
}

template <typename TI>
__global__ __launch_bounds__(256) void k_pack_thr(
    const TI* __restrict__ x, const float* __restrict__ thr,
    int8_t* __restrict__ out, long total, int C) {
  long i = (long)blockIdx.x * blockDim.x + threadIdx.x;
  if (i >= total) return;
  float v = (float)x[i];
  float th = thr[(int)(i % C)];
  out[i] = (v > th) ? (int8_t)1 : ((v < th) ? (int8_t)-1 : (int8_t)0);
}

__global__ __launch_bounds__(256) void k_gemm1(
    const int8_t* __restrict__ W,
    const int8_t* __restrict__ S,
    int16_t* __restrict__ Y) {
  const int T = T_LEN, C = B_CH, O = H_CH;
  int lane = threadIdx.x & 31;
  int wave = threadIdx.x >> 5;
  int n = blockIdx.z;
  int o0 = blockIdx.y * 16;
  int m = lane & 15;
  int koff = (lane & 16) ? 8 : 0;
  int trow = lane & 15;
  int khalf = (lane >> 4) & 1;
  int tcol = lane & 15;
  int ohi = (lane & 16) ? 8 : 0;

  v8i a0 = load_afrag(W, C, o0 + m, 0,  koff);
  v8i a1 = load_afrag(W, C, o0 + m, 64, koff);

  const int8_t* sn = S + (size_t)n * T * C;
  int16_t* yn = Y + (size_t)n * T * O;
  const int step = gridDim.x * 8;

  for (int dt = blockIdx.x * 8 + wave; dt < T / 64; dt += step) {
    int t0 = dt * 64;
    const int8_t* sb = sn + (size_t)t0 * C;
    __builtin_prefetch(sb + (size_t)step * 64 * C + (size_t)trow * C, 0, 1);

    v8i b0[4], b1[4], acc[4];
#pragma unroll
    for (int j = 0; j < 4; ++j)
      b0[j] = load_bfrag(sb + (size_t)(16 * j) * C, C, trow, khalf);
#pragma unroll
    for (int j = 0; j < 4; ++j)
      b1[j] = load_bfrag(sb + (size_t)(16 * j) * C + 64, C, trow, khalf);
#pragma unroll
    for (int j = 0; j < 4; ++j) {
      v8i z = {0, 0, 0, 0, 0, 0, 0, 0};
      acc[j] = __builtin_amdgcn_wmma_i32_16x16x64_iu8(true, a0, true, b0[j], z, false, false);
    }
#pragma unroll
    for (int j = 0; j < 4; ++j)
      acc[j] = __builtin_amdgcn_wmma_i32_16x16x64_iu8(true, a1, true, b1[j], acc[j], false, false);

#pragma unroll
    for (int j = 0; j < 4; ++j) {
      int16_t* yp = yn + (size_t)(t0 + 16 * j + tcol) * O + o0 + ohi;
      int4 pk;
      pk.x = (acc[j][0] & 0xffff) | (acc[j][1] << 16);
      pk.y = (acc[j][2] & 0xffff) | (acc[j][3] << 16);
      pk.z = (acc[j][4] & 0xffff) | (acc[j][5] << 16);
      pk.w = (acc[j][6] & 0xffff) | (acc[j][7] << 16);
      *(int4*)(yp) = pk;
    }
  }
}

__global__ __launch_bounds__(256) void k_dwconv(
    const int8_t* __restrict__ S,
    const int8_t* __restrict__ Wd,
    int8_t* __restrict__ D, int dil) {
  const int T = T_LEN, C = H_CH;
  long total = (long)N_BATCH * T * (C / 4);
  long i = (long)blockIdx.x * blockDim.x + threadIdx.x;
  if (i >= total) return;
  int c4 = (int)(i & (C / 4 - 1));
  long nt = i >> 7;
  int t = (int)(nt % T);
  int n = (int)(nt / T);
  int c0 = c4 * 4;
  const int8_t* base = S + ((size_t)n * T) * C + c0;
  int mid = *(const int*)(base + (size_t)t * C);
  int lo = (t - dil >= 0) ? *(const int*)(base + (size_t)(t - dil) * C) : 0;
  int hi = (t + dil < T)  ? *(const int*)(base + (size_t)(t + dil) * C) : 0;
  int out = 0;
#pragma unroll
  for (int j = 0; j < 4; ++j) {
    int w0 = Wd[(c0 + j) * 3 + 0];
    int w1 = Wd[(c0 + j) * 3 + 1];
    int w2 = Wd[(c0 + j) * 3 + 2];
    int x0 = (int)(int8_t)(lo  >> (8 * j));
    int x1 = (int)(int8_t)(mid >> (8 * j));
    int x2 = (int)(int8_t)(hi  >> (8 * j));
    int r = w0 * x0 + w1 * x1 + w2 * x2;
    out |= (r & 0xff) << (8 * j);
  }
  *(int*)(D + ((size_t)n * T + t) * C + c0) = out;
}

__global__ __launch_bounds__(256) void k_gemm2(
    const int8_t* __restrict__ W,
    const int8_t* __restrict__ S,
    float* __restrict__ R) {
  const int T = T_LEN, C = H_CH, O = B_CH;
  int lane = threadIdx.x & 31;
  int wave = threadIdx.x >> 5;
  int n = blockIdx.z;
  int o0 = blockIdx.y * 16;
  int m = lane & 15;
  int koff = (lane & 16) ? 8 : 0;
  int trow = lane & 15;
  int khalf = (lane >> 4) & 1;
  int tcol = lane & 15;
  int ohi = (lane & 16) ? 8 : 0;

  v8i a[8];
#pragma unroll
  for (int ks = 0; ks < 8; ++ks)
    a[ks] = load_afrag(W, C, o0 + m, ks * 64, koff);

  const int8_t* sn = S + (size_t)n * T * C;
  float* rn = R + (size_t)n * T * O;
  const int step = gridDim.x * 8;

  for (int dt = blockIdx.x * 8 + wave; dt < T / 64; dt += step) {
    int t0 = dt * 64;
    const int8_t* sb = sn + (size_t)t0 * C;
    __builtin_prefetch(sb + (size_t)step * 64 * C + (size_t)trow * C, 0, 1);

    v8i acc[4], bt[4], bn[4];
#pragma unroll
    for (int j = 0; j < 4; ++j) {
      v8i z = {0, 0, 0, 0, 0, 0, 0, 0};
      acc[j] = z;
      bt[j] = load_bfrag(sb + (size_t)(16 * j) * C, C, trow, khalf);
    }
#pragma unroll
    for (int ks = 0; ks < 8; ++ks) {
      if (ks < 7) {
#pragma unroll
        for (int j = 0; j < 4; ++j)
          bn[j] = load_bfrag(sb + (size_t)(16 * j) * C + (ks + 1) * 64, C, trow, khalf);
      }
#pragma unroll
      for (int j = 0; j < 4; ++j)
        acc[j] = __builtin_amdgcn_wmma_i32_16x16x64_iu8(true, a[ks], true, bt[j], acc[j], false, false);
#pragma unroll
      for (int j = 0; j < 4; ++j)
        bt[j] = bn[j];
    }
#pragma unroll
    for (int j = 0; j < 4; ++j) {
      float* rp = rn + (size_t)(t0 + 16 * j + tcol) * O + o0 + ohi;
      float4 u0 = *(float4*)(rp);
      float4 u1 = *(float4*)(rp + 4);
      u0.x += (float)acc[j][0]; u0.y += (float)acc[j][1];
      u0.z += (float)acc[j][2]; u0.w += (float)acc[j][3];
      u1.x += (float)acc[j][4]; u1.y += (float)acc[j][5];
      u1.z += (float)acc[j][6]; u1.w += (float)acc[j][7];
      *(float4*)(rp)     = u0;
      *(float4*)(rp + 4) = u1;
    }
  }
}

extern "C" void kernel_launch(void* const* d_in, const int* in_sizes, int n_in,
                              void* d_out, int out_size, void* d_ws, size_t ws_size,
                              hipStream_t stream) {
  (void)in_sizes; (void)n_in; (void)out_size; (void)ws_size;
  const float* x  = (const float*)d_in[0];
  const float* W1 = (const float*)d_in[1];
  const float* Wd = (const float*)d_in[2];
  const float* W2 = (const float*)d_in[3];
  const float* g1 = (const float*)d_in[4];
  const float* b1 = (const float*)d_in[5];
  const float* g2 = (const float*)d_in[6];
  const float* b2 = (const float*)d_in[7];
  const float* g3 = (const float*)d_in[8];
  const float* b3 = (const float*)d_in[9];

  const int N = N_BATCH, B = B_CH, H = H_CH, T = T_LEN, BLOCKS = 8;

  char* ws = (char*)d_ws;
  size_t off = 0;
  auto carve = [&](size_t bytes) -> char* {
    char* p = ws + off;
    off = (off + bytes + 255) & ~(size_t)255;
    return p;
  };
  float*   resid = (float*)carve((size_t)N * T * B * 4);
  int8_t*  s1  = (int8_t*)carve((size_t)N * T * B);
  int16_t* y1  = (int16_t*)carve((size_t)N * T * H * 2);
  int8_t*  s2  = (int8_t*)carve((size_t)N * T * H);
  int8_t*  dd  = (int8_t*)carve((size_t)N * T * H);
  int8_t*  s3  = (int8_t*)carve((size_t)N * T * H);
  int8_t*  sw1 = (int8_t*)carve((size_t)BLOCKS * H * B);
  int8_t*  swd = (int8_t*)carve((size_t)BLOCKS * H * 3);
  int8_t*  sw2 = (int8_t*)carve((size_t)BLOCKS * B * H);
  float* thr1 = (float*)carve((size_t)B * 4);
  float* thr2 = (float*)carve((size_t)H * 4);
  float* thr3 = (float*)carve((size_t)H * 4);

  long n1 = (long)N * T * B;
  long n2 = (long)N * T * H;
  long n3 = n2 / 4;

  k_t_in<<<(int)((n1 + 255) / 256), 256, 0, stream>>>(x, resid);

  long nw1 = (long)BLOCKS * H * B;
  long nwd = (long)BLOCKS * H * 3;
  long nw2 = (long)BLOCKS * B * H;
  k_sign<<<(int)((nw1 + 255) / 256), 256, 0, stream>>>(W1, sw1, nw1);
  k_sign<<<(int)((nwd + 255) / 256), 256, 0, stream>>>(Wd, swd, nwd);
  k_sign<<<(int)((nw2 + 255) / 256), 256, 0, stream>>>(W2, sw2, nw2);

  for (int i = 0; i < BLOCKS; ++i) {
    int dil = 1 << i;
    k_stats_ntc<float><<<B, 256, 0, stream>>>(resid, g1 + i * B, b1 + i * B, thr1, B, (long)N * T);
    k_pack_thr<float><<<(int)((n1 + 255) / 256), 256, 0, stream>>>(resid, thr1, s1, n1, B);
    k_gemm1<<<dim3(16, H / 16, N), 256, 0, stream>>>(sw1 + (size_t)i * H * B, s1, y1);
    k_stats_ntc<int16_t><<<H, 256, 0, stream>>>(y1, g2 + i * H, b2 + i * H, thr2, H, (long)N * T);
    k_pack_thr<int16_t><<<(int)((n2 + 255) / 256), 256, 0, stream>>>(y1, thr2, s2, n2, H);
    k_dwconv<<<(int)((n3 + 255) / 256), 256, 0, stream>>>(s2, swd + (size_t)i * H * 3, dd, dil);
    k_stats_ntc<int8_t><<<H, 256, 0, stream>>>(dd, g3 + i * H, b3 + i * H, thr3, H, (long)N * T);
    k_pack_thr<int8_t><<<(int)((n2 + 255) / 256), 256, 0, stream>>>(dd, thr3, s3, n2, H);
    k_gemm2<<<dim3(16, B / 16, N), 256, 0, stream>>>(sw2 + (size_t)i * B * H, s3, resid);
  }

  k_t_out<<<(int)((n1 + 255) / 256), 256, 0, stream>>>(resid, (float*)d_out);
}
